// MultiheadAttention_377957122745
// MI455X (gfx1250) — matrix operand order repaired and checked by reading, not compiled, not run on hardware
//
#include <hip/hip_runtime.h>
#include <hip/hip_bf16.h>
#include <cstdint>

typedef __attribute__((ext_vector_type(16))) __bf16 v16bf;
typedef __attribute__((ext_vector_type(8)))  __bf16 v8bf;
typedef __attribute__((ext_vector_type(8)))  float  v8f;
typedef __attribute__((ext_vector_type(4)))  float  v4f;
typedef __attribute__((ext_vector_type(4)))  unsigned v4u;
typedef __attribute__((ext_vector_type(8)))  int    v8i;
typedef __attribute__((ext_vector_type(4)))  int    v4i;

#define CH     128
#define HW     4096
#define WIDTH  64
#define NBATCH 4
#define NPIX   (NBATCH * HW)   // 16384
#define K2     25
#define NHEAD  4
#define DHEAD  32

__device__ __forceinline__ __bf16 f2bf(float f) {
  unsigned u = __builtin_bit_cast(unsigned, f);
  unsigned r = u + 0x7FFFu + ((u >> 16) & 1u);   // round-to-nearest-even
  unsigned short h = (unsigned short)(r >> 16);
  return __builtin_bit_cast(__bf16, h);
}

// ---------------------------------------------------------------------------
// 1) weights fp32 -> bf16 (row-major [o][c], the WMMA-A source layout)
// ---------------------------------------------------------------------------
__global__ void convert_weights_kernel(const float* __restrict__ Wq,
                                       const float* __restrict__ Wk,
                                       const float* __restrict__ Wv,
                                       const float* __restrict__ Wfc,
                                       __bf16* __restrict__ out) {
  int i = blockIdx.x * blockDim.x + threadIdx.x;            // 0 .. 65535
  if (i >= 4 * CH * CH) return;
  const float* src = (i < CH*CH) ? Wq : (i < 2*CH*CH) ? Wk : (i < 3*CH*CH) ? Wv : Wfc;
  out[i] = f2bf(src[i & (CH*CH - 1)]);
}

// ---------------------------------------------------------------------------
// 2) q/k/v (n,c,hw) fp32 -> pixel-major (n,hw,c) bf16
// ---------------------------------------------------------------------------
__global__ void xpose_bf16_kernel(const float* __restrict__ q,
                                  const float* __restrict__ k,
                                  const float* __restrict__ v,
                                  __bf16* __restrict__ qb,
                                  __bf16* __restrict__ kb,
                                  __bf16* __restrict__ vb) {
  int idx = blockIdx.x * blockDim.x + threadIdx.x;          // global pixel 0..16383
  if (idx >= NPIX) return;
  const float* s; __bf16* d;
  if (blockIdx.y == 0)      { s = q; d = qb; }
  else if (blockIdx.y == 1) { s = k; d = kb; }
  else                      { s = v; d = vb; }
  int n = idx >> 12, p = idx & (HW - 1);
  const float* sp = s + (size_t)n * CH * HW + p;
  __bf16* dp = d + (size_t)idx * CH;
#pragma unroll 8
  for (int c = 0; c < CH; ++c)
    dp[c] = f2bf(sp[(size_t)c * HW]);
}

// ---------------------------------------------------------------------------
// WMMA fragment loaders (ISA 7.12.2 16-bit layouts, wave32)
// A 16x32 (M=o,K=c): lane l -> M = l&15 ; elems 0..7 = K (l<16?0:8)..+7,
//                                     elems 8..15 = K 16+(l<16?0:8)..+7
// B 32x16 (K=c,N=p): lane l -> N = l&15 ; elems 0..7 = K (l<16?0:8)..+7,
//                                     elems 8..15 = K 16+(l<16?0:8)..+7
// ---------------------------------------------------------------------------
__device__ __forceinline__ v16bf load_a_frag(const __bf16* W,
                                             int otile, int kb, int lane) {
  int row = otile + (lane & 15);
  int kh  = (lane & 16) ? 8 : 0;
  const __bf16* p = W + row * CH + kb + kh;
  v8bf lo = *(const v8bf*)p;          // LDS-resident -> ds_load_b128
  v8bf hi = *(const v8bf*)(p + 16);
  v16bf f;
#pragma unroll
  for (int i = 0; i < 8; ++i) { f[i] = lo[i]; f[i + 8] = hi[i]; }
  return f;
}

__device__ __forceinline__ v16bf load_b_frag(const __bf16* __restrict__ X,
                                             int col0, int kb, int lane) {
  int col   = col0 + (lane & 15);
  int kbase = kb + ((lane & 16) ? 8 : 0);
  const __bf16* p = X + (size_t)col * CH + kbase;
  v8bf lo = *(const v8bf*)p;
  v8bf hi = *(const v8bf*)(p + 16);
  v16bf f;
#pragma unroll
  for (int i = 0; i < 8; ++i) { f[i] = lo[i]; f[i + 8] = hi[i]; }
  return f;
}

// ---------------------------------------------------------------------------
// 3/5) GEMM:  Out[n][o][p] = sum_c W[o][c] * X[n][p][c]
//   W staged into LDS once per block via the Tensor Data Mover (TDM).
//   One wave owns a 32(M) x 64(N) strip: 8 accumulators, each B fragment
//   feeds two WMMAs; B fragments batch-loaded per K-step so WMMAs issue
//   back-to-back instead of waiting per-load.
//   CHMAJOR=false -> Out pixel-major fp32 (n,hw,c)   (projections)
//   CHMAJOR=true  -> Out channel-major fp32 (n,c,hw) (final 'nohw' output)
// ---------------------------------------------------------------------------
template <bool CHMAJOR>
__global__ void __launch_bounds__(256)
gemm_wmma_kernel(const __bf16* __restrict__ W,
                 const __bf16* __restrict__ X,
                 float* __restrict__ Out) {
  __shared__ __bf16 lds_w[CH * CH];   // 32 KB staged weight matrix

  int wave = threadIdx.x >> 5;
  int lane = threadIdx.x & 31;

  // ---- TDM: stage all of W (128x128 bf16) into LDS; one wave issues it.
  if (wave == 0) {
    unsigned lds_addr = (unsigned)(uintptr_t)(&lds_w[0]);   // LDS byte offset
    unsigned long long ga = (unsigned long long)(uintptr_t)W;
    // D# group 0: count=1 | lds_addr | global_addr[56:0] | type=2
    v4u g0 = { 1u,
               lds_addr,
               (unsigned)(ga & 0xFFFFFFFFu),
               (unsigned)((ga >> 32) & 0x01FFFFFFu) | (2u << 30) };
    // D# group 1: data_size=2B, tensor 128x128, tile 128x128, stride 128
    v8i g1 = { (int)(1u << 16),    // [17:16] data_size=1 (2B); wg_mask=0
               (int)(128u << 16),  // [63:48] tensor_dim0 lo16 = 128
               (int)(128u << 16),  // [95:80] tensor_dim1 lo16 = 128
               (int)(128u << 16),  // [127:112] tile_dim0 = 128
               (int)(128u),        // [143:128] tile_dim1 = 128; tile_dim2=0
               (int)(128u),        // [191:160] tensor_dim0_stride = 128
               0, 0 };
    v4i z4 = {0, 0, 0, 0};
#if __clang_major__ >= 23
    v8i z8 = {0, 0, 0, 0, 0, 0, 0, 0};
    __builtin_amdgcn_tensor_load_to_lds(g0, g1, z4, z4, z8, 0);
#else
    __builtin_amdgcn_tensor_load_to_lds(g0, g1, z4, z4, 0);
#endif
    __builtin_amdgcn_s_wait_tensorcnt(0);
  }
  __syncthreads();

  // ---- wave -> output strip mapping: 4(n) x 4(mt) x 64(pg) = 1024 waves
  int wid = blockIdx.x * 8 + wave;
  int n     = wid >> 8;
  int rem   = wid & 255;
  int otile = (rem >> 6) << 5;        // 0..96 step 32
  int p0    = (rem & 63) << 6;        // 0..4032 step 64
  const __bf16* Xn = X + (size_t)n * HW * CH;

  v8f acc0[4] = {};
  v8f acc1[4] = {};
#pragma unroll
  for (int kb = 0; kb < CH; kb += 32) {
    v16bf a0 = load_a_frag(lds_w, otile,      kb, lane);
    v16bf a1 = load_a_frag(lds_w, otile + 16, kb, lane);
    v16bf b[4];
#pragma unroll
    for (int t = 0; t < 4; ++t)
      b[t] = load_b_frag(Xn, p0 + t * 16, kb, lane);
#pragma unroll
    for (int t = 0; t < 4; ++t) {
      acc0[t] = __builtin_amdgcn_wmma_f32_16x16x32_bf16(
          false, a0, false, b[t], (short)0, acc0[t], false, false);
      acc1[t] = __builtin_amdgcn_wmma_f32_16x16x32_bf16(
          false, a1, false, b[t], (short)0, acc1[t], false, false);
    }
  }

  // ---- store: lane l holds N = p0+t*16+(l&15); rows o = om+8*(l>>4)+r
  int pl = lane & 15;
  int oh = (lane & 16) ? 8 : 0;
#pragma unroll
  for (int m = 0; m < 2; ++m) {
    int ob = otile + m * 16 + oh;
    if (!CHMAJOR) {
      float* base = Out + (size_t)n * HW * CH;
#pragma unroll
      for (int t = 0; t < 4; ++t) {
        v8f acc = m ? acc1[t] : acc0[t];
        int p = p0 + t * 16 + pl;
        float* dst = base + (size_t)p * CH + ob;       // 8 contiguous floats
        v4f lo = { acc[0], acc[1], acc[2], acc[3] };
        v4f hi = { acc[4], acc[5], acc[6], acc[7] };
        *(v4f*)dst       = lo;
        *(v4f*)(dst + 4) = hi;
      }
    } else {
      float* base = Out + (size_t)n * CH * HW;
#pragma unroll
      for (int t = 0; t < 4; ++t) {
        v8f acc = m ? acc1[t] : acc0[t];
        int p = p0 + t * 16 + pl;
#pragma unroll
        for (int r = 0; r < 8; ++r)
          base[(size_t)(ob + r) * HW + p] = acc[r];
      }
    }
  }
}

// ---------------------------------------------------------------------------
// 4) fused flow->grid + bilinear(zero-pad) sample + softmax + AV
//    one wave per pixel; lane = d within head; heads looped.
// ---------------------------------------------------------------------------
__global__ void attn_kernel(const float* __restrict__ qp,   // (n,hw,c) fp32
                            const float* __restrict__ kp,
                            const float* __restrict__ vp,
                            const float* __restrict__ flow, // (n,2,hw)
                            float* __restrict__ attn_out,   // (n,4,25,hw)
                            __bf16* __restrict__ aout) {    // (n,hw,c) bf16
  int wid  = blockIdx.x * (blockDim.x >> 5) + (threadIdx.x >> 5);
  int lane = threadIdx.x & 31;
  if (wid >= NPIX) return;
  int n = wid >> 12, p = wid & (HW - 1);
  int y = p >> 6,    x = p & (WIDTH - 1);

  float fx = flow[(size_t)n * 2 * HW + p];
  float fy = flow[(size_t)n * 2 * HW + HW + p];
  float xb = (float)x + fx;        // sample x = xb + sx
  float yb = (float)y + fy;

  const float* qpix = qp + (size_t)wid * CH;
  const float inv_sqrt_d = 0.17677669529663689f;  // 1/sqrt(32)

  for (int head = 0; head < NHEAD; ++head) {
    int cbase = head * DHEAD + lane;
    float qv = qpix[cbase] * inv_sqrt_d;

    float logits[K2];
    float vs[K2];
#pragma unroll
    for (int s = 0; s < K2; ++s) {
      float xs = xb + (float)(s % 5 - 2);
      float ys = yb + (float)(s / 5 - 2);
      float x0f = floorf(xs), y0f = floorf(ys);
      float wx = xs - x0f,    wy = ys - y0f;
      int   x0 = (int)x0f,    y0 = (int)y0f;
      float ka = 0.f, va = 0.f;
#pragma unroll
      for (int t = 0; t < 4; ++t) {
        int xi = x0 + (t & 1);
        int yi = y0 + (t >> 1);
        float wt = ((t & 1) ? wx : 1.f - wx) * ((t >> 1) ? wy : 1.f - wy);
        if (xi >= 0 && xi < WIDTH && yi >= 0 && yi < WIDTH) {
          size_t off = ((size_t)(n * HW + yi * WIDTH + xi)) * CH + cbase;
          ka += wt * kp[off];
          va += wt * vp[off];
        }
      }
      vs[s] = va;
      float lg = qv * ka;
#pragma unroll
      for (int m = 16; m > 0; m >>= 1)       // butterfly: all lanes get sum
        lg += __shfl_xor(lg, m, 32);
      logits[s] = lg;
    }

    // softmax over 25 (replicated across lanes)
    float mx = -3.402823466e38f;
#pragma unroll
    for (int s = 0; s < K2; ++s) mx = fmaxf(mx, logits[s]);
    float sum = 0.f, myattn = 0.f;
#pragma unroll
    for (int s = 0; s < K2; ++s) {
      float e = __expf(logits[s] - mx);
      logits[s] = e;
      sum += e;
      if (lane == s) myattn = e;             // lane s keeps attn[s]
    }
    float inv = 1.f / sum;

    if (lane < K2)
      attn_out[(size_t)n * NHEAD * K2 * HW + (size_t)head * K2 * HW
               + (size_t)lane * HW + p] = myattn * inv;

    float od = 0.f;
#pragma unroll
    for (int s = 0; s < K2; ++s) od += logits[s] * vs[s];
    od *= inv;
    aout[(size_t)wid * CH + cbase] = f2bf(od);
  }
}

// ---------------------------------------------------------------------------
extern "C" void kernel_launch(void* const* d_in, const int* in_sizes, int n_in,
                              void* d_out, int out_size, void* d_ws, size_t ws_size,
                              hipStream_t stream) {
  const float* q    = (const float*)d_in[0];
  const float* k    = (const float*)d_in[1];
  const float* v    = (const float*)d_in[2];
  const float* flow = (const float*)d_in[3];
  const float* Wq   = (const float*)d_in[4];
  const float* Wk   = (const float*)d_in[5];
  const float* Wv   = (const float*)d_in[6];
  const float* Wfc  = (const float*)d_in[7];
  float* out = (float*)d_out;

  // workspace carve-out
  char* ws = (char*)d_ws;
  size_t off = 0;
  __bf16* qbf = (__bf16*)(ws + off); off += (size_t)NPIX * CH * 2;
  __bf16* kbf = (__bf16*)(ws + off); off += (size_t)NPIX * CH * 2;
  __bf16* vbf = (__bf16*)(ws + off); off += (size_t)NPIX * CH * 2;
  __bf16* wbf = (__bf16*)(ws + off); off += (size_t)4 * CH * CH * 2;
  float*  qp  = (float*)(ws + off);  off += (size_t)NPIX * CH * 4;
  float*  kp  = (float*)(ws + off);  off += (size_t)NPIX * CH * 4;
  float*  vp  = (float*)(ws + off);  off += (size_t)NPIX * CH * 4;
  __bf16* aout = (__bf16*)(ws + off); off += (size_t)NPIX * CH * 2;

  __bf16* Wqbf  = wbf;
  __bf16* Wkbf  = wbf + CH * CH;
  __bf16* Wvbf  = wbf + 2 * CH * CH;
  __bf16* Wfcbf = wbf + 3 * CH * CH;

  // 1) weights -> bf16
  convert_weights_kernel<<<(4 * CH * CH + 255) / 256, 256, 0, stream>>>(
      Wq, Wk, Wv, Wfc, wbf);

  // 2) inputs -> pixel-major bf16
  dim3 xg(NPIX / 256, 3);
  xpose_bf16_kernel<<<xg, 256, 0, stream>>>(q, k, v, qbf, kbf, vbf);

  // 3) projections (WMMA + TDM-staged weights), pixel-major fp32 outputs
  const int gemm_blocks = (NBATCH * 4 * 64) / 8;   // 1024 waves / 8 per block
  gemm_wmma_kernel<false><<<gemm_blocks, 256, 0, stream>>>(Wqbf, qbf, qp);
  gemm_wmma_kernel<false><<<gemm_blocks, 256, 0, stream>>>(Wkbf, kbf, kp);
  gemm_wmma_kernel<false><<<gemm_blocks, 256, 0, stream>>>(Wvbf, vbf, vp);

  // 4) fused sampling + attention
  float* attn_part = out + (size_t)NPIX * CH;      // after 'out' tensor
  attn_kernel<<<NPIX / 8, 256, 0, stream>>>(qp, kp, vp, flow, attn_part, aout);

  // 5) final projection (WMMA), channel-major 'nohw' straight into d_out
  gemm_wmma_kernel<true><<<gemm_blocks, 256, 0, stream>>>(Wfcbf, aout, out);
}
